// TransformerBlock_1116691497151
// MI455X (gfx1250) — matrix operand order repaired and checked by reading, not compiled, not run on hardware
//
#include <hip/hip_runtime.h>
#include <cstdint>
#include <cstddef>

// ---------------------------------------------------------------------------
// Types / helpers
// ---------------------------------------------------------------------------
typedef unsigned short u16;
typedef unsigned int   u32;
typedef __attribute__((ext_vector_type(16))) __bf16 bf16x16;
typedef __attribute__((ext_vector_type(8)))  float  f32x8;

struct alignas(16) U4 { u32 x, y, z, w; };
union FragU { bf16x16 v; U4 q[2]; };

#define DEV static __device__ __forceinline__

constexpr int Bc = 2, Sc = 2048, Dc = 1024;
constexpr int Hc = 16, HKVc = 4, HDc = 64, Gc = Hc / HKVc;
constexpr int Ec = 8, TOPKc = 2, HIDc = 2752;
constexpr int MT = Bc * Sc;                    // 4096 tokens

DEV u16 f2bf(float f) {
  u32 u = __float_as_uint(f);
  u32 r = u + 0x7FFFu + ((u >> 16) & 1u);      // round-to-nearest-even
  return (u16)(r >> 16);
}

DEV f32x8 wmma_bf16(bf16x16 a, bf16x16 b, f32x8 c) {
  // D = A(16x32 bf16) * B(32x16 bf16) + C(16x16 f32)
  return __builtin_amdgcn_wmma_f32_16x16x32_bf16(false, a, false, b,
                                                 (short)0, c, false, false);
}

// A fragment: 16x32 bf16, row-major source with `stride` (u16 units).
// lanes 0-15: M=lane, K = {0..7, 16..23}; lanes 16-31: M=lane-16, K = {8..15, 24..31}
DEV bf16x16 fragA(const u16* base, int stride) {
  int l = threadIdx.x & 31;
  const u16* p = base + (l & 15) * stride + ((l >> 4) << 3);
  FragU f;
  f.q[0] = *(const U4*)(p);
  f.q[1] = *(const U4*)(p + 16);
  return f.v;
}

// B fragment: 32x16 bf16 from B^T stored row-major (rows = N, cols = K).
// lanes 0-15: N=lane, K = {0..7, 16..23}; lanes 16-31: N=lane-16, K = {8..15, 24..31}
DEV bf16x16 fragB(const u16* base, int stride) {
  int l = threadIdx.x & 31;
  const u16* p = base + (l & 15) * stride + ((l >> 4) << 3);
  FragU f;
  f.q[0] = *(const U4*)(p);
  f.q[1] = *(const U4*)(p + 16);
  return f.v;
}

// ---------------------------------------------------------------------------
// 1) Weight convert + transpose: src [batch][K][N] f32 -> dst [batch][N][K] bf16
// ---------------------------------------------------------------------------
__global__ __launch_bounds__(256) void k_w_transpose(
    const float* __restrict__ src, u16* __restrict__ dst,
    int K, int N, long sStride, long dStride) {
  __shared__ float tile[32][33];
  long sb = (long)blockIdx.z * sStride;
  long db = (long)blockIdx.z * dStride;
  int k0 = blockIdx.y * 32, n0 = blockIdx.x * 32;
  int tx = threadIdx.x & 31, ty = threadIdx.x >> 5;   // 32 x 8
#pragma unroll
  for (int i = 0; i < 4; ++i) {
    int k = k0 + ty + i * 8, n = n0 + tx;
    tile[ty + i * 8][tx] = (k < K && n < N) ? src[sb + (long)k * N + n] : 0.f;
  }
  __syncthreads();
#pragma unroll
  for (int i = 0; i < 4; ++i) {
    int n = n0 + ty + i * 8, k = k0 + tx;
    if (n < N && k < K) dst[db + (long)n * K + k] = f2bf(tile[tx][ty + i * 8]);
  }
}

// ---------------------------------------------------------------------------
// 2) RMSNorm (f32 in -> bf16 out), one block per token row
// ---------------------------------------------------------------------------
__global__ __launch_bounds__(256) void k_rmsnorm(
    const float* __restrict__ x, const float* __restrict__ w,
    u16* __restrict__ out) {
  int row = blockIdx.x;
  const float* xr = x + (long)row * Dc;
  __shared__ float red[256];
  float s = 0.f;
  for (int d = threadIdx.x; d < Dc; d += 256) { float v = xr[d]; s += v * v; }
  red[threadIdx.x] = s; __syncthreads();
  for (int o = 128; o > 0; o >>= 1) {
    if ((int)threadIdx.x < o) red[threadIdx.x] += red[threadIdx.x + o];
    __syncthreads();
  }
  float rs = rsqrtf(red[0] / (float)Dc + 1e-5f);
  for (int d = threadIdx.x; d < Dc; d += 256)
    out[(long)row * Dc + d] = f2bf(xr[d] * rs * w[d]);
}

// ---------------------------------------------------------------------------
// 3) Generic WMMA GEMM: C[M,N] = A[M,K](bf16) * Bt[N,K](bf16) (+ resid), f32 out
//    block tile 128x64, 8 waves (4x2), each wave 32x32 (2x2 WMMA tiles)
// ---------------------------------------------------------------------------
__global__ __launch_bounds__(256) void k_gemm_bf16(
    const u16* __restrict__ A, const u16* __restrict__ Bt,
    float* __restrict__ C, const float* __restrict__ resid,
    int M, int N, int K) {
  __shared__ __align__(16) u16 sA[128 * 40];
  __shared__ __align__(16) u16 sB[64 * 40];
  int m0 = blockIdx.y * 128, n0 = blockIdx.x * 64;
  int wave = threadIdx.x >> 5;
  int wm = (wave & 3) * 32, wn = (wave >> 2) * 32;
  int l = threadIdx.x & 31, ln = l & 15, hi = l >> 4;

  f32x8 c[2][2] = {};
  int nk = K >> 5;
  for (int kc = 0; kc < nk; ++kc) {
    __syncthreads();
#pragma unroll
    for (int i = 0; i < 2; ++i) {                 // A: 128x32 = 512 U4
      int id = threadIdx.x + i * 256;
      int r = id >> 2, cq = (id & 3) * 8;
      *(U4*)&sA[r * 40 + cq] =
          *(const U4*)(A + (long)(m0 + r) * K + kc * 32 + cq);
    }
    {                                             // B: 64x32 = 256 U4
      int r = threadIdx.x >> 2, cq = (threadIdx.x & 3) * 8;
      *(U4*)&sB[r * 40 + cq] =
          *(const U4*)(Bt + (long)(n0 + r) * K + kc * 32 + cq);
      if (kc + 1 < nk)                            // global_prefetch_b8
        __builtin_prefetch(Bt + (long)(n0 + r) * K + (kc + 1) * 32 + cq, 0, 1);
    }
    __syncthreads();
    bf16x16 a0 = fragA(sA + (wm)      * 40, 40);
    bf16x16 a1 = fragA(sA + (wm + 16) * 40, 40);
    bf16x16 b0 = fragB(sB + (wn)      * 40, 40);
    bf16x16 b1 = fragB(sB + (wn + 16) * 40, 40);
    c[0][0] = wmma_bf16(a0, b0, c[0][0]);
    c[0][1] = wmma_bf16(a0, b1, c[0][1]);
    c[1][0] = wmma_bf16(a1, b0, c[1][0]);
    c[1][1] = wmma_bf16(a1, b1, c[1][1]);
  }
#pragma unroll
  for (int i = 0; i < 2; ++i)
#pragma unroll
    for (int j = 0; j < 2; ++j)
#pragma unroll
      for (int v = 0; v < 8; ++v) {
        int m = m0 + wm + i * 16 + v + hi * 8;
        int n = n0 + wn + j * 16 + ln;
        float val = c[i][j][v];
        if (resid) val += resid[(long)m * N + n];
        C[(long)m * N + n] = val;
      }
}

// ---------------------------------------------------------------------------
// 4) RoPE + head-major bf16 pack for Q, K; bf16 pack for V
// ---------------------------------------------------------------------------
__global__ __launch_bounds__(256) void k_rope_pack(
    const float* __restrict__ q, const float* __restrict__ k,
    const float* __restrict__ v, const float* __restrict__ fc,
    const float* __restrict__ fs, u16* __restrict__ qb,
    u16* __restrict__ kb, u16* __restrict__ vb) {
  int tok = blockIdx.x;
  int b = tok / Sc, s = tok % Sc;
  for (int i = threadIdx.x; i < Hc * (HDc / 2); i += 256) {
    int h = i >> 5, d2 = i & 31;
    float tr = q[(long)tok * Dc + h * HDc + 2 * d2];
    float ti = q[(long)tok * Dc + h * HDc + 2 * d2 + 1];
    float cc = fc[s * 32 + d2], ss = fs[s * 32 + d2];
    long o = ((long)(b * Hc + h) * Sc + s) * HDc + 2 * d2;
    qb[o]     = f2bf(tr * cc - ti * ss);
    qb[o + 1] = f2bf(tr * ss + ti * cc);
  }
  for (int i = threadIdx.x; i < HKVc * (HDc / 2); i += 256) {
    int h = i >> 5, d2 = i & 31;
    float tr = k[(long)tok * (HKVc * HDc) + h * HDc + 2 * d2];
    float ti = k[(long)tok * (HKVc * HDc) + h * HDc + 2 * d2 + 1];
    float cc = fc[s * 32 + d2], ss = fs[s * 32 + d2];
    long o = ((long)(b * HKVc + h) * Sc + s) * HDc + 2 * d2;
    kb[o]     = f2bf(tr * cc - ti * ss);
    kb[o + 1] = f2bf(tr * ss + ti * cc);
  }
  for (int i = threadIdx.x; i < HKVc * HDc; i += 256) {
    int h = i / HDc, d = i % HDc;
    vb[((long)(b * HKVc + h) * Sc + s) * HDc + d] =
        f2bf(v[(long)tok * (HKVc * HDc) + i]);
  }
}

// ---------------------------------------------------------------------------
// 5) Flash attention (causal, GQA). grid (S/128, B*H), 8 waves x 16 q-rows.
//    Scores and PV via WMMA bf16; online softmax in f32.
// ---------------------------------------------------------------------------
__global__ __launch_bounds__(256) void k_flash_attn(
    const u16* __restrict__ qb, const u16* __restrict__ kb,
    const u16* __restrict__ vb, u16* __restrict__ attn) {
  __shared__ __align__(16) u16 sK[32 * 72];      // K tile [kv][hd]
  __shared__ __align__(16) u16 sV[64 * 40];      // V^T tile [hd][kv]
  __shared__ __align__(16) u16 sP[8][16 * 40];   // per-wave P tile [q][kv]
  int bh = blockIdx.y, b = bh / Hc, h = bh % Hc, hk = h / Gc;
  int wave = threadIdx.x >> 5, l = threadIdx.x & 31, ln = l & 15, hi = l >> 4;
  int q0 = blockIdx.x * 128 + wave * 16;

  const u16* qrow = qb + ((long)(b * Hc + h) * Sc + q0) * HDc;
  bf16x16 qf0 = fragA(qrow, HDc);
  bf16x16 qf1 = fragA(qrow + 32, HDc);

  f32x8 o[4] = {};
  float mrow[8], lrow[8];
#pragma unroll
  for (int v = 0; v < 8; ++v) { mrow[v] = -3.0e38f; lrow[v] = 0.f; }
  const float scale = 0.125f;                    // 1/sqrt(64)

  const u16* Kg = kb + (long)(b * HKVc + hk) * Sc * HDc;
  const u16* Vg = vb + (long)(b * HKVc + hk) * Sc * HDc;
  int ntiles = blockIdx.x * 4 + 4;               // causal upper bound for block

  for (int t = 0; t < ntiles; ++t) {
    int k0 = t * 32;
    __syncthreads();
    {   // stage K 32x64 (one U4 per thread)
      int r = threadIdx.x >> 3, cc = (threadIdx.x & 7) * 8;
      *(U4*)&sK[r * 72 + cc] = *(const U4*)(Kg + (long)(k0 + r) * HDc + cc);
    }
    {   // stage V^T 64x32 (8 u16 per thread, transposed store)
      int e0 = threadIdx.x * 8;
      int r = e0 >> 6, c0 = e0 & 63;
      const u16* src = Vg + (long)(k0 + r) * HDc + c0;
#pragma unroll
      for (int i = 0; i < 8; ++i) sV[(c0 + i) * 40 + r] = src[i];
    }
    __syncthreads();

    if (k0 <= q0 + 15) {                         // wave-uniform causal guard
      f32x8 s0 = {}, s1 = {};
      s0 = wmma_bf16(qf0, fragB(sK, 72), s0);
      s0 = wmma_bf16(qf1, fragB(sK + 32, 72), s0);
      s1 = wmma_bf16(qf0, fragB(sK + 16 * 72, 72), s1);
      s1 = wmma_bf16(qf1, fragB(sK + 16 * 72 + 32, 72), s1);

      float pm[8];
#pragma unroll
      for (int v = 0; v < 8; ++v) {
        int m = q0 + v + hi * 8;
        int c0i = k0 + ln, c1i = c0i + 16;
        float x0 = s0[v] * scale + ((c0i <= m) ? 0.f : -1.0e9f);
        float x1 = s1[v] * scale + ((c1i <= m) ? 0.f : -1.0e9f);
        s0[v] = x0; s1[v] = x1;
        pm[v] = fmaxf(x0, x1);
      }
#pragma unroll
      for (int v = 0; v < 8; ++v) {              // row max across 16 lanes
        float x = pm[v];
        x = fmaxf(x, __shfl_xor(x, 1, 32));
        x = fmaxf(x, __shfl_xor(x, 2, 32));
        x = fmaxf(x, __shfl_xor(x, 4, 32));
        x = fmaxf(x, __shfl_xor(x, 8, 32));
        pm[v] = x;
      }
#pragma unroll
      for (int v = 0; v < 8; ++v) {
        float mnew = fmaxf(mrow[v], pm[v]);
        float alpha = __expf(mrow[v] - mnew);
        mrow[v] = mnew;
        float p0 = __expf(s0[v] - mnew), p1 = __expf(s1[v] - mnew);
        float ps = p0 + p1;
        ps += __shfl_xor(ps, 1, 32);
        ps += __shfl_xor(ps, 2, 32);
        ps += __shfl_xor(ps, 4, 32);
        ps += __shfl_xor(ps, 8, 32);
        lrow[v] = lrow[v] * alpha + ps;
#pragma unroll
        for (int dt = 0; dt < 4; ++dt) o[dt][v] *= alpha;
        sP[wave][(v + hi * 8) * 40 + ln]      = f2bf(p0);
        sP[wave][(v + hi * 8) * 40 + ln + 16] = f2bf(p1);
      }
      // wave-private LDS round trip: DS ops are in-order per wave
      asm volatile("s_wait_dscnt 0" ::: "memory");
      bf16x16 pf = fragA(&sP[wave][0], 40);
#pragma unroll
      for (int dt = 0; dt < 4; ++dt)
        o[dt] = wmma_bf16(pf, fragB(&sV[dt * 16 * 40], 40), o[dt]);
    }
  }
#pragma unroll
  for (int dt = 0; dt < 4; ++dt)
#pragma unroll
    for (int v = 0; v < 8; ++v) {
      int m = q0 + v + hi * 8;
      float val = o[dt][v] / lrow[v];
      attn[(long)(b * Sc + m) * Dc + h * HDc + dt * 16 + ln] = f2bf(val);
    }
}

// ---------------------------------------------------------------------------
// 6) MoE routing: fused ffn RMSNorm + gate logits + top-2 softmax
// ---------------------------------------------------------------------------
__global__ __launch_bounds__(256) void k_route(
    const float* __restrict__ oa, const float* __restrict__ wn,
    const float* __restrict__ gate, u16* __restrict__ h2b,
    int* __restrict__ sel, float* __restrict__ wtt) {
  int tok = blockIdx.x;
  __shared__ float red[256];
  __shared__ float h2s[Dc];
  __shared__ float lg[Ec];
  const float* xr = oa + (long)tok * Dc;
  float s = 0.f;
  for (int d = threadIdx.x; d < Dc; d += 256) { float v = xr[d]; s += v * v; }
  red[threadIdx.x] = s; __syncthreads();
  for (int o = 128; o > 0; o >>= 1) {
    if ((int)threadIdx.x < o) red[threadIdx.x] += red[threadIdx.x + o];
    __syncthreads();
  }
  float rs = rsqrtf(red[0] / (float)Dc + 1e-5f);
  for (int d = threadIdx.x; d < Dc; d += 256) {
    float hv = xr[d] * rs * wn[d];
    h2s[d] = hv;
    h2b[(long)tok * Dc + d] = f2bf(hv);
  }
  __syncthreads();
  float acc[Ec];
#pragma unroll
  for (int e = 0; e < Ec; ++e) acc[e] = 0.f;
  for (int d = threadIdx.x; d < Dc; d += 256) {
    float hv = h2s[d];
#pragma unroll
    for (int e = 0; e < Ec; ++e) acc[e] += hv * gate[d * Ec + e];
  }
  for (int e = 0; e < Ec; ++e) {
    red[threadIdx.x] = acc[e]; __syncthreads();
    for (int o = 128; o > 0; o >>= 1) {
      if ((int)threadIdx.x < o) red[threadIdx.x] += red[threadIdx.x + o];
      __syncthreads();
    }
    if (threadIdx.x == 0) lg[e] = red[0];
    __syncthreads();
  }
  if (threadIdx.x == 0) {
    int i0 = 0;
    for (int e = 1; e < Ec; ++e) if (lg[e] > lg[i0]) i0 = e;
    int i1 = -1;
    for (int e = 0; e < Ec; ++e)
      if (e != i0 && (i1 < 0 || lg[e] > lg[i1])) i1 = e;
    float e1 = __expf(lg[i1] - lg[i0]);
    float inv = 1.f / (1.f + e1);
    sel[tok * 2] = i0;  sel[tok * 2 + 1] = i1;
    wtt[tok * 2] = inv; wtt[tok * 2 + 1] = e1 * inv;
  }
}

// 7) histogram + exclusive scan over experts (single block)
__global__ __launch_bounds__(256) void k_scan(
    const int* __restrict__ sel, int* __restrict__ cnt,
    int* __restrict__ segoff, int* __restrict__ cursor) {
  __shared__ int lc[Ec];
  if (threadIdx.x < Ec) lc[threadIdx.x] = 0;
  __syncthreads();
  for (int i = threadIdx.x; i < MT * TOPKc; i += 256) atomicAdd(&lc[sel[i]], 1);
  __syncthreads();
  if (threadIdx.x == 0) {
    int off = 0;
    for (int e = 0; e < Ec; ++e) {
      segoff[e] = off; cursor[e] = off; cnt[e] = lc[e]; off += lc[e];
    }
  }
}

// 8) slot assignment (compact per-expert token lists)
__global__ __launch_bounds__(256) void k_assign(
    const int* __restrict__ sel, int* __restrict__ cursor,
    int* __restrict__ idx, int* __restrict__ slots) {
  int i = blockIdx.x * 256 + threadIdx.x;
  if (i >= MT) return;
#pragma unroll
  for (int j = 0; j < TOPKc; ++j) {
    int e = sel[i * 2 + j];
    int slot = atomicAdd(&cursor[e], 1);
    idx[slot] = i;
    slots[i * 2 + j] = slot;
  }
}

// ---------------------------------------------------------------------------
// 9) MoE MLP stage 1: act = silu(h2 @ W1t) * (h2 @ W3t), gathered rows
// ---------------------------------------------------------------------------
__global__ __launch_bounds__(256) void k_moe_mlp1(
    const u16* __restrict__ h2b, const u16* __restrict__ w1t,
    const u16* __restrict__ w3t, const int* __restrict__ idx,
    const int* __restrict__ segoff, const int* __restrict__ cnt,
    u16* __restrict__ act) {
  int e = blockIdx.z;
  int ce = cnt[e], so = segoff[e];
  int m0 = blockIdx.y * 128;
  if (m0 >= ce) return;
  int n0 = blockIdx.x * 64;
  __shared__ __align__(16) u16 sA[128 * 40];
  __shared__ __align__(16) u16 sB1[64 * 40];
  __shared__ __align__(16) u16 sB3[64 * 40];
  __shared__ int rowTok[128];
  if (threadIdx.x < 128) {
    int r = m0 + threadIdx.x;
    rowTok[threadIdx.x] = (r < ce) ? idx[so + r] : -1;
  }
  const u16* W1 = w1t + (long)e * HIDc * Dc;
  const u16* W3 = w3t + (long)e * HIDc * Dc;
  int wave = threadIdx.x >> 5;
  int wm = (wave & 3) * 32, wn = (wave >> 2) * 32;
  int l = threadIdx.x & 31, ln = l & 15, hi = l >> 4;

  f32x8 c1[2][2] = {}, c3[2][2] = {};
  for (int kc = 0; kc < Dc / 32; ++kc) {
    __syncthreads();
#pragma unroll
    for (int i = 0; i < 2; ++i) {
      int id = threadIdx.x + i * 256;
      int r = id >> 2, cq = (id & 3) * 8;
      int tk = rowTok[r];
      U4 val = {0, 0, 0, 0};
      if (tk >= 0) val = *(const U4*)(h2b + (long)tk * Dc + kc * 32 + cq);
      *(U4*)&sA[r * 40 + cq] = val;
    }
    {
      int r = threadIdx.x >> 2, cq = (threadIdx.x & 3) * 8;
      *(U4*)&sB1[r * 40 + cq] =
          *(const U4*)(W1 + (long)(n0 + r) * Dc + kc * 32 + cq);
      *(U4*)&sB3[r * 40 + cq] =
          *(const U4*)(W3 + (long)(n0 + r) * Dc + kc * 32 + cq);
    }
    __syncthreads();
    bf16x16 a0 = fragA(sA + (wm)      * 40, 40);
    bf16x16 a1 = fragA(sA + (wm + 16) * 40, 40);
    bf16x16 p0 = fragB(sB1 + (wn)      * 40, 40);
    bf16x16 p1 = fragB(sB1 + (wn + 16) * 40, 40);
    bf16x16 r0 = fragB(sB3 + (wn)      * 40, 40);
    bf16x16 r1 = fragB(sB3 + (wn + 16) * 40, 40);
    c1[0][0] = wmma_bf16(a0, p0, c1[0][0]);
    c1[0][1] = wmma_bf16(a0, p1, c1[0][1]);
    c1[1][0] = wmma_bf16(a1, p0, c1[1][0]);
    c1[1][1] = wmma_bf16(a1, p1, c1[1][1]);
    c3[0][0] = wmma_bf16(a0, r0, c3[0][0]);
    c3[0][1] = wmma_bf16(a0, r1, c3[0][1]);
    c3[1][0] = wmma_bf16(a1, r0, c3[1][0]);
    c3[1][1] = wmma_bf16(a1, r1, c3[1][1]);
  }
#pragma unroll
  for (int i = 0; i < 2; ++i)
#pragma unroll
    for (int j = 0; j < 2; ++j)
#pragma unroll
      for (int v = 0; v < 8; ++v) {
        int r = m0 + wm + i * 16 + v + hi * 8;
        if (r < ce) {
          int n = n0 + wn + j * 16 + ln;
          float g = c1[i][j][v], u = c3[i][j][v];
          float sig = 1.f / (1.f + __expf(-g));
          act[(long)(so + r) * HIDc + n] = f2bf(g * sig * u);
        }
      }
}

// ---------------------------------------------------------------------------
// 10) MoE MLP stage 2: eout = act @ W2t (per-slot rows, f32 out)
// ---------------------------------------------------------------------------
__global__ __launch_bounds__(256) void k_moe_mlp2(
    const u16* __restrict__ act, const u16* __restrict__ w2t,
    const int* __restrict__ segoff, const int* __restrict__ cnt,
    float* __restrict__ eout) {
  int e = blockIdx.z;
  int ce = cnt[e], so = segoff[e];
  int m0 = blockIdx.y * 128;
  if (m0 >= ce) return;
  int n0 = blockIdx.x * 64;
  __shared__ __align__(16) u16 sA[128 * 40];
  __shared__ __align__(16) u16 sB[64 * 40];
  const u16* Aex = act + (long)so * HIDc;
  const u16* W = w2t + (long)e * Dc * HIDc;
  int wave = threadIdx.x >> 5;
  int wm = (wave & 3) * 32, wn = (wave >> 2) * 32;
  int l = threadIdx.x & 31, ln = l & 15, hi = l >> 4;

  f32x8 c[2][2] = {};
  for (int kc = 0; kc < HIDc / 32; ++kc) {
    __syncthreads();
#pragma unroll
    for (int i = 0; i < 2; ++i) {
      int id = threadIdx.x + i * 256;
      int r = id >> 2, cq = (id & 3) * 8;
      U4 val = {0, 0, 0, 0};
      if (m0 + r < ce)
        val = *(const U4*)(Aex + (long)(m0 + r) * HIDc + kc * 32 + cq);
      *(U4*)&sA[r * 40 + cq] = val;
    }
    {
      int r = threadIdx.x >> 2, cq = (threadIdx.x & 3) * 8;
      *(U4*)&sB[r * 40 + cq] =
          *(const U4*)(W + (long)(n0 + r) * HIDc + kc * 32 + cq);
    }
    __syncthreads();
    bf16x16 a0 = fragA(sA + (wm)      * 40, 40);
    bf16x16 a1 = fragA(sA + (wm + 16) * 40, 40);
    bf16x16 b0 = fragB(sB + (wn)      * 40, 40);
    bf16x16 b1 = fragB(sB + (wn + 16) * 40, 40);
    c[0][0] = wmma_bf16(a0, b0, c[0][0]);
    c[0][1] = wmma_bf16(a0, b1, c[0][1]);
    c[1][0] = wmma_bf16(a1, b0, c[1][0]);
    c[1][1] = wmma_bf16(a1, b1, c[1][1]);
  }
#pragma unroll
  for (int i = 0; i < 2; ++i)
#pragma unroll
    for (int j = 0; j < 2; ++j)
#pragma unroll
      for (int v = 0; v < 8; ++v) {
        int r = m0 + wm + i * 16 + v + hi * 8;
        if (r < ce) {
          int n = n0 + wn + j * 16 + ln;
          eout[(long)(so + r) * Dc + n] = c[i][j][v];
        }
      }
}

// 11) deterministic per-token combine: out = oa + w0*eout[slot0] + w1*eout[slot1]
__global__ __launch_bounds__(256) void k_combine(
    const float* __restrict__ oa, const float* __restrict__ eout,
    const int* __restrict__ slots, const float* __restrict__ wtt,
    float* __restrict__ out) {
  long i = (long)blockIdx.x * 256 + threadIdx.x;
  if (i >= (long)MT * Dc) return;
  int tok = (int)(i / Dc), d = (int)(i % Dc);
  int s0 = slots[tok * 2], s1 = slots[tok * 2 + 1];
  float f = wtt[tok * 2]     * eout[(long)s0 * Dc + d] +
            wtt[tok * 2 + 1] * eout[(long)s1 * Dc + d];
  out[i] = oa[i] + f;
}

// ---------------------------------------------------------------------------
// Host launch
// ---------------------------------------------------------------------------
extern "C" void kernel_launch(void* const* d_in, const int* in_sizes, int n_in,
                              void* d_out, int out_size, void* d_ws,
                              size_t ws_size, hipStream_t stream) {
  const float* x   = (const float*)d_in[0];
  const float* fc  = (const float*)d_in[2];
  const float* fs  = (const float*)d_in[3];
  const float* anw = (const float*)d_in[5];
  const float* fnw = (const float*)d_in[6];
  const float* wq  = (const float*)d_in[7];
  const float* wk  = (const float*)d_in[8];
  const float* wv  = (const float*)d_in[9];
  const float* wo  = (const float*)d_in[10];
  const float* gw  = (const float*)d_in[11];
  const float* w1  = (const float*)d_in[12];
  const float* w2  = (const float*)d_in[13];
  const float* w3  = (const float*)d_in[14];
  float* out = (float*)d_out;

  char* wsb = (char*)d_ws;
  size_t off = 0;
  auto alloc = [&](size_t bytes) -> void* {
    void* p = wsb + off;
    off = (off + bytes + 255) & ~(size_t)255;
    return p;
  };
  u16* wq_t  = (u16*)alloc((size_t)Dc * Dc * 2);
  u16* wk_t  = (u16*)alloc((size_t)(HKVc * HDc) * Dc * 2);
  u16* wv_t  = (u16*)alloc((size_t)(HKVc * HDc) * Dc * 2);
  u16* wo_t  = (u16*)alloc((size_t)Dc * Dc * 2);
  u16* w1_t  = (u16*)alloc((size_t)Ec * HIDc * Dc * 2);
  u16* w3_t  = (u16*)alloc((size_t)Ec * HIDc * Dc * 2);
  u16* w2_t  = (u16*)alloc((size_t)Ec * Dc * HIDc * 2);
  u16* hb    = (u16*)alloc((size_t)MT * Dc * 2);
  float* qf  = (float*)alloc((size_t)MT * Dc * 4);
  float* kf  = (float*)alloc((size_t)MT * HKVc * HDc * 4);
  float* vf  = (float*)alloc((size_t)MT * HKVc * HDc * 4);
  u16* qbh   = (u16*)alloc((size_t)Bc * Hc * Sc * HDc * 2);
  u16* kbh   = (u16*)alloc((size_t)Bc * HKVc * Sc * HDc * 2);
  u16* vbh   = (u16*)alloc((size_t)Bc * HKVc * Sc * HDc * 2);
  u16* attnb = (u16*)alloc((size_t)MT * Dc * 2);
  float* oa  = (float*)alloc((size_t)MT * Dc * 4);
  u16* h2b   = (u16*)alloc((size_t)MT * Dc * 2);
  int* sel   = (int*)alloc((size_t)MT * 2 * 4);
  float* wtt = (float*)alloc((size_t)MT * 2 * 4);
  int* cnt   = (int*)alloc(Ec * 4);
  int* segof = (int*)alloc(Ec * 4);
  int* curs  = (int*)alloc(Ec * 4);
  int* idxl  = (int*)alloc((size_t)MT * 2 * 4);
  int* slots = (int*)alloc((size_t)MT * 2 * 4);
  u16* act   = (u16*)alloc((size_t)MT * 2 * HIDc * 2);
  float* eou = (float*)alloc((size_t)MT * 2 * Dc * 4);
  (void)ws_size; (void)in_sizes; (void)n_in; (void)out_size;

  dim3 blk(256);
  // weight convert + transpose to bf16 [N][K]
  k_w_transpose<<<dim3(32, 32, 1), blk, 0, stream>>>(wq, wq_t, Dc, Dc, 0, 0);
  k_w_transpose<<<dim3(8, 32, 1),  blk, 0, stream>>>(wk, wk_t, Dc, HKVc * HDc, 0, 0);
  k_w_transpose<<<dim3(8, 32, 1),  blk, 0, stream>>>(wv, wv_t, Dc, HKVc * HDc, 0, 0);
  k_w_transpose<<<dim3(32, 32, 1), blk, 0, stream>>>(wo, wo_t, Dc, Dc, 0, 0);
  k_w_transpose<<<dim3(HIDc / 32, Dc / 32, Ec), blk, 0, stream>>>(
      w1, w1_t, Dc, HIDc, (long)Dc * HIDc, (long)HIDc * Dc);
  k_w_transpose<<<dim3(HIDc / 32, Dc / 32, Ec), blk, 0, stream>>>(
      w3, w3_t, Dc, HIDc, (long)Dc * HIDc, (long)HIDc * Dc);
  k_w_transpose<<<dim3(Dc / 32, HIDc / 32, Ec), blk, 0, stream>>>(
      w2, w2_t, HIDc, Dc, (long)HIDc * Dc, (long)Dc * HIDc);

  // attention path
  k_rmsnorm<<<MT, blk, 0, stream>>>(x, anw, hb);
  k_gemm_bf16<<<dim3(Dc / 64, MT / 128), blk, 0, stream>>>(
      hb, wq_t, qf, nullptr, MT, Dc, Dc);
  k_gemm_bf16<<<dim3((HKVc * HDc) / 64, MT / 128), blk, 0, stream>>>(
      hb, wk_t, kf, nullptr, MT, HKVc * HDc, Dc);
  k_gemm_bf16<<<dim3((HKVc * HDc) / 64, MT / 128), blk, 0, stream>>>(
      hb, wv_t, vf, nullptr, MT, HKVc * HDc, Dc);
  k_rope_pack<<<MT, blk, 0, stream>>>(qf, kf, vf, fc, fs, qbh, kbh, vbh);
  k_flash_attn<<<dim3(Sc / 128, Bc * Hc), blk, 0, stream>>>(qbh, kbh, vbh, attnb);
  k_gemm_bf16<<<dim3(Dc / 64, MT / 128), blk, 0, stream>>>(
      attnb, wo_t, oa, x, MT, Dc, Dc);     // + residual x -> out_attention

  // MoE path
  k_route<<<MT, blk, 0, stream>>>(oa, fnw, gw, h2b, sel, wtt);
  k_scan<<<1, blk, 0, stream>>>(sel, cnt, segof, curs);
  k_assign<<<MT / 256, blk, 0, stream>>>(sel, curs, idxl, slots);
  k_moe_mlp1<<<dim3(HIDc / 64, MT / 128, Ec), blk, 0, stream>>>(
      h2b, w1_t, w3_t, idxl, segof, cnt, act);
  k_moe_mlp2<<<dim3(Dc / 64, MT / 128, Ec), blk, 0, stream>>>(
      act, w2_t, segof, cnt, eou);
  k_combine<<<(MT * Dc) / 256, blk, 0, stream>>>(oa, eou, slots, wtt, out);
}
